// MultiHeadAttention_29996051595826
// MI455X (gfx1250) — matrix operand order repaired and checked by reading, not compiled, not run on hardware
//
#include <hip/hip_runtime.h>
#include <hip/hip_bf16.h>
#include <math.h>

typedef __bf16 bf16_t;
typedef __attribute__((ext_vector_type(16))) __bf16 v16bf;
typedef __attribute__((ext_vector_type(8)))  __bf16 v8bf;
typedef __attribute__((ext_vector_type(8)))  float  v8f;

static constexpr int Bn = 4, Tn = 2048, Cn = 1024, Hn = 16, Dn = 64;

// ---------- WMMA fragment helpers (CDNA5 16x16x32 bf16, wave32) ----------
// A (16xK, 16-bit): lane holds row M=lane%16; half=lane/16 selects K-chunk:
//   elements 0..15 -> K = 16*half + 0..15
__device__ __forceinline__ v16bf load_a_frag(const bf16_t* rowp, int half) {
  return *(const v16bf*)(rowp + 16 * half);
}
// B (Kx16, 16-bit) in (N,K)-major storage: lane holds column N=lane%16,
// 16 contiguous K values starting at 16*half.
__device__ __forceinline__ v16bf load_b_frag(const bf16_t* rowp, int half) {
  return *(const v16bf*)(rowp + 16 * half);
}
__device__ __forceinline__ v8f wmma_bf16(v16bf a, v16bf b, v8f c) {
  return __builtin_amdgcn_wmma_f32_16x16x32_bf16(false, a, false, b, (short)0, c,
                                                 false, false);
}
// reductions across the 16-lane half-groups (rows live in lanes 0-15 / 16-31)
__device__ __forceinline__ float rmax16(float v) {
  #pragma unroll
  for (int m = 1; m < 16; m <<= 1) v = fmaxf(v, __shfl_xor(v, m, 32));
  return v;
}
__device__ __forceinline__ float rsum16(float v) {
  #pragma unroll
  for (int m = 1; m < 16; m <<= 1) v += __shfl_xor(v, m, 32);
  return v;
}

// ---------- prep: fp32 -> bf16 converts ----------
__global__ void cvt_f32_bf16(const float* __restrict__ src, bf16_t* __restrict__ dst, int n) {
  int i = blockIdx.x * blockDim.x + threadIdx.x;
  if (i < n) dst[i] = (bf16_t)src[i];
}
// src[h][c][d] (H,C,D) -> dst[h][d][c]  ((N,K)-major for WMMA B operand)
__global__ void cvt_transpose_w(const float* __restrict__ src, bf16_t* __restrict__ dst, int n) {
  int i = blockIdx.x * blockDim.x + threadIdx.x;
  if (i >= n) return;
  int h = i / (Cn * Dn);
  int r = i % (Cn * Dn);
  int c = r / Dn;
  int d = r % Dn;
  dst[((size_t)h * Dn + d) * Cn + c] = (bf16_t)src[i];
}

// ---------- QKV projection ----------
// blockIdx.y selects Q / K / V. Per wave: one 32x64 output tile (two 16-row
// M-tiles sharing every weight B-fragment -> 2 WMMAs per B load, 8 accs only).
__global__ void qkv_gemm_kernel(const bf16_t* __restrict__ X,
                                const bf16_t* __restrict__ Wqt,
                                const bf16_t* __restrict__ Wkt,
                                const bf16_t* __restrict__ Wvt,
                                bf16_t* __restrict__ Q, bf16_t* __restrict__ K,
                                bf16_t* __restrict__ Vt) {
  int which = blockIdx.y;
  const bf16_t* W = (which == 0) ? Wqt : (which == 1) ? Wkt : Wvt;
  bf16_t* Out     = (which == 0) ? Q   : (which == 1) ? K   : Vt;

  int gid = blockIdx.x * (blockDim.x >> 5) + (threadIdx.x >> 5);  // 0..4095
  int bh  = gid >> 6;            // (b*H+h); 64 M-pairs per (b,h)
  int m0  = (gid & 63) * 32;
  int b   = bh >> 4;
  int h   = bh & 15;
  int lane16 = threadIdx.x & 15;
  int half   = (threadIdx.x >> 4) & 1;

  v8f acc0[4], acc1[4];
  v8f z = {};
  #pragma unroll
  for (int i = 0; i < 4; ++i) { acc0[i] = z; acc1[i] = z; }

  const bf16_t* xrow0 = X + ((size_t)b * Tn + m0 + lane16) * Cn;
  const bf16_t* xrow1 = xrow0 + (size_t)16 * Cn;
  const bf16_t* wlane = W + ((size_t)h * Dn + lane16) * Cn;

  for (int kc = 0; kc < Cn; kc += 32) {
    v16bf a0 = load_a_frag(xrow0 + kc, half);
    v16bf a1 = load_a_frag(xrow1 + kc, half);
    #pragma unroll
    for (int nt = 0; nt < 4; ++nt) {
      v16bf bfr = load_b_frag(wlane + (size_t)nt * 16 * Cn + kc, half);
      acc0[nt] = wmma_bf16(a0, bfr, acc0[nt]);
      acc1[nt] = wmma_bf16(a1, bfr, acc1[nt]);
    }
  }
  size_t obase = (size_t)bh * Tn * Dn;  // Q,K: [T,D]; V: [D,T]
  if (which != 2) {
    #pragma unroll
    for (int nt = 0; nt < 4; ++nt) {
      int d = nt * 16 + lane16;
      #pragma unroll
      for (int r = 0; r < 8; ++r) {
        int t = m0 + r + 8 * half;
        Out[obase + (size_t)t * Dn + d]        = (bf16_t)acc0[nt][r];
        Out[obase + (size_t)(t + 16) * Dn + d] = (bf16_t)acc1[nt][r];
      }
    }
  } else {
    #pragma unroll
    for (int nt = 0; nt < 4; ++nt) {
      int d = nt * 16 + lane16;
      #pragma unroll
      for (int r = 0; r < 8; ++r) {
        int t = m0 + r + 8 * half;
        Out[obase + (size_t)d * Tn + t]      = (bf16_t)acc0[nt][r];
        Out[obase + (size_t)d * Tn + t + 16] = (bf16_t)acc1[nt][r];
      }
    }
  }
}

// ---------- flash attention: per wave one 16-row query tile ----------
__global__ void attn_kernel(const bf16_t* __restrict__ Q,
                            const bf16_t* __restrict__ K,
                            const bf16_t* __restrict__ Vt,
                            bf16_t* __restrict__ OH) {
  __shared__ __align__(32) bf16_t ptile[4][16 * 32];
  const int T16 = Tn / 16;
  int wv_id = threadIdx.x >> 5;
  int gid = blockIdx.x * (blockDim.x >> 5) + wv_id;
  int b   = gid / (Hn * T16);
  int rem = gid % (Hn * T16);
  int h   = rem / T16;
  int m0  = (rem % T16) * 16;
  int lane16 = threadIdx.x & 15;
  int half   = (threadIdx.x >> 4) & 1;
  const float scale = 0.03125f;  // C**-0.5, C = 1024 (n_embd scaling, per source)

  const bf16_t* qrow  = Q  + ((size_t)(b * Hn + h) * Tn + m0 + lane16) * Dn;
  const bf16_t* kbase = K  + (size_t)(b * Hn + h) * Tn * Dn;
  const bf16_t* vbase = Vt + (size_t)(b * Hn + h) * Dn * Tn;
  bf16_t* myP = ptile[wv_id];

  v16bf aq0 = load_a_frag(qrow, half);        // d = 0..31
  v16bf aq1 = load_a_frag(qrow + 32, half);   // d = 32..63

  v8f o[4]; v8f z = {};
  #pragma unroll
  for (int i = 0; i < 4; ++i) o[i] = z;
  float m_r[8], l_r[8];
  #pragma unroll
  for (int r = 0; r < 8; ++r) { m_r[r] = -__builtin_inff(); l_r[r] = 0.f; }

  for (int s0 = 0; s0 < m0 + 16; s0 += 32) {
    // scores: two 16x16 tiles covering keys [s0, s0+32)
    v8f S0 = z, S1 = z;
    const bf16_t* k0 = kbase + (size_t)(s0 + lane16) * Dn;
    const bf16_t* k1 = kbase + (size_t)(s0 + 16 + lane16) * Dn;
    S0 = wmma_bf16(aq0, load_b_frag(k0, half), S0);
    S0 = wmma_bf16(aq1, load_b_frag(k0 + 32, half), S0);
    S1 = wmma_bf16(aq0, load_b_frag(k1, half), S1);
    S1 = wmma_bf16(aq1, load_b_frag(k1 + 32, half), S1);

    // causal mask + online softmax (f32), P tile staged to LDS for A-layout
    #pragma unroll
    for (int r = 0; r < 8; ++r) {
      int tq = m0 + r + 8 * half;
      float x0 = (s0 + lane16) <= tq      ? S0[r] * scale : -__builtin_inff();
      float x1 = (s0 + 16 + lane16) <= tq ? S1[r] * scale : -__builtin_inff();
      float rm   = rmax16(fmaxf(x0, x1));
      float mnew = fmaxf(m_r[r], rm);
      float alpha = __expf(m_r[r] - mnew);
      float p0 = __expf(x0 - mnew);
      float p1 = __expf(x1 - mnew);
      l_r[r] = l_r[r] * alpha + rsum16(p0 + p1);
      m_r[r] = mnew;
      o[0][r] *= alpha; o[1][r] *= alpha; o[2][r] *= alpha; o[3][r] *= alpha;
      int row = r + 8 * half;
      myP[row * 32 + lane16]      = (bf16_t)p0;
      myP[row * 32 + 16 + lane16] = (bf16_t)p1;
    }
    asm volatile("s_wait_dscnt 0" ::: "memory");  // LDS store -> load ordering
    v16bf ap = load_a_frag(myP + lane16 * 32, half);  // P as 16x32 A-fragment
    #pragma unroll
    for (int nt = 0; nt < 4; ++nt) {
      const bf16_t* vrow = vbase + (size_t)(nt * 16 + lane16) * Tn + s0;
      o[nt] = wmma_bf16(ap, load_b_frag(vrow, half), o[nt]);
    }
  }
  // normalize and store concat-head output [B,T,C] bf16
  #pragma unroll
  for (int r = 0; r < 8; ++r) {
    float inv = 1.0f / l_r[r];
    int t = m0 + r + 8 * half;
    #pragma unroll
    for (int nt = 0; nt < 4; ++nt) {
      OH[((size_t)b * Tn + t) * Cn + h * Dn + nt * 16 + lane16] =
          (bf16_t)(o[nt][r] * inv);
    }
  }
}

// ---------- output projection: Y = OH @ Wproj^T + b, f32 out ----------
// Per wave: 32x64 tile (two M-tiles sharing each Wproj B-fragment).
__global__ void proj_kernel(const bf16_t* __restrict__ OH,
                            const bf16_t* __restrict__ Wp,  // [C,C] row-major = (N,K)
                            const float* __restrict__ bias,
                            float* __restrict__ out) {
  int gid = blockIdx.x * (blockDim.x >> 5) + (threadIdx.x >> 5);  // 0..4095
  int mp = gid >> 4;          // 256 row-pair tiles over B*T
  int n0 = (gid & 15) * 64;   // 16 column groups of 64
  int m0 = mp * 32;
  int lane16 = threadIdx.x & 15;
  int half   = (threadIdx.x >> 4) & 1;

  v8f acc0[4], acc1[4];
  v8f z = {};
  #pragma unroll
  for (int i = 0; i < 4; ++i) { acc0[i] = z; acc1[i] = z; }

  const bf16_t* arow0 = OH + (size_t)(m0 + lane16) * Cn;
  const bf16_t* arow1 = arow0 + (size_t)16 * Cn;
  const bf16_t* wlane = Wp + (size_t)(n0 + lane16) * Cn;

  for (int kc = 0; kc < Cn; kc += 32) {
    v16bf a0 = load_a_frag(arow0 + kc, half);
    v16bf a1 = load_a_frag(arow1 + kc, half);
    #pragma unroll
    for (int nt = 0; nt < 4; ++nt) {
      v16bf bfr = load_b_frag(wlane + (size_t)nt * 16 * Cn + kc, half);
      acc0[nt] = wmma_bf16(a0, bfr, acc0[nt]);
      acc1[nt] = wmma_bf16(a1, bfr, acc1[nt]);
    }
  }
  #pragma unroll
  for (int nt = 0; nt < 4; ++nt) {
    float bb = bias[n0 + nt * 16 + lane16];
    #pragma unroll
    for (int r = 0; r < 8; ++r) {
      int row = m0 + r + 8 * half;
      out[(size_t)row * Cn + n0 + nt * 16 + lane16]        = acc0[nt][r] + bb;
      out[(size_t)(row + 16) * Cn + n0 + nt * 16 + lane16] = acc1[nt][r] + bb;
    }
  }
}

extern "C" void kernel_launch(void* const* d_in, const int* in_sizes, int n_in,
                              void* d_out, int out_size, void* d_ws, size_t ws_size,
                              hipStream_t stream) {
  const float* x     = (const float*)d_in[0];
  const float* Wq    = (const float*)d_in[1];
  const float* Wk    = (const float*)d_in[2];
  const float* Wv    = (const float*)d_in[3];
  const float* Wproj = (const float*)d_in[4];
  const float* bproj = (const float*)d_in[5];

  char* ws = (char*)d_ws;
  const size_t SZ_X   = (size_t)Bn * Tn * Cn * 2;       // 16 MB
  const size_t SZ_W   = (size_t)Hn * Dn * Cn * 2;       //  2 MB
  const size_t SZ_WP  = (size_t)Cn * Cn * 2;            //  2 MB
  const size_t SZ_QKV = (size_t)Bn * Hn * Tn * Dn * 2;  // 16 MB
  size_t off = 0;
  bf16_t* Xb  = (bf16_t*)(ws + off); off += SZ_X;
  bf16_t* Wqt = (bf16_t*)(ws + off); off += SZ_W;
  bf16_t* Wkt = (bf16_t*)(ws + off); off += SZ_W;
  bf16_t* Wvt = (bf16_t*)(ws + off); off += SZ_W;
  bf16_t* Wpb = (bf16_t*)(ws + off); off += SZ_WP;
  bf16_t* Qb  = (bf16_t*)(ws + off); off += SZ_QKV;
  bf16_t* Kb  = (bf16_t*)(ws + off); off += SZ_QKV;
  bf16_t* Vtb = (bf16_t*)(ws + off); off += SZ_QKV;
  bf16_t* OHb = (bf16_t*)(ws + off); off += SZ_X;

  int nx = Bn * Tn * Cn;
  cvt_f32_bf16<<<(nx + 255) / 256, 256, 0, stream>>>(x, Xb, nx);
  int nw = Hn * Cn * Dn;
  cvt_transpose_w<<<(nw + 255) / 256, 256, 0, stream>>>(Wq, Wqt, nw);
  cvt_transpose_w<<<(nw + 255) / 256, 256, 0, stream>>>(Wk, Wkt, nw);
  cvt_transpose_w<<<(nw + 255) / 256, 256, 0, stream>>>(Wv, Wvt, nw);
  int np = Cn * Cn;
  cvt_f32_bf16<<<(np + 255) / 256, 256, 0, stream>>>(Wproj, Wpb, np);

  // QKV: 4096 waves per matrix (32x64 tiles), y-dim selects Q/K/V
  qkv_gemm_kernel<<<dim3(1024, 3), 128, 0, stream>>>(Xb, Wqt, Wkt, Wvt, Qb, Kb, Vtb);
  // attention: 8192 waves, one 16-row query tile each
  attn_kernel<<<2048, 128, 0, stream>>>(Qb, Kb, Vtb, OHb);
  // projection: 4096 waves (32x64 tiles)
  proj_kernel<<<1024, 128, 0, stream>>>(OHb, Wpb, bproj, (float*)d_out);
}
